// TGATSamplerModel_10479720202341
// MI455X (gfx1250) — compile-verified, matrix operand order checked by reading, not run on hardware
//
#include <hip/hip_runtime.h>
#include <math.h>

// ---------------------------------------------------------------------------
// TGAT-style temporal graph transformer for MI455X (gfx1250, wave32, WMMA).
// Dense GEMMs (K=128) use v_wmma_f32_16x16x32_f16, f32 accumulate. Operands
// are kept f16-resident: activations stored f16, weights pre-transposed to
// f16 [N,128] so every WMMA fragment is two contiguous 16B b128 loads.
// Each wave owns a 16x64 tile (4 accumulators, shared A fragment).
// Edge time-encoding GEMM fuses cos(t*freq+phase) into the A-fragment.
// Segment ops via L2 atomics; edge encoding stored fp16.
// ---------------------------------------------------------------------------

typedef _Float16 f16;
typedef __attribute__((ext_vector_type(16))) _Float16 v16h;
typedef __attribute__((ext_vector_type(8)))  _Float16 v8h;
typedef __attribute__((ext_vector_type(8)))  float    v8f;

#define HID   128
#define HEADS 4
#define CH    32
#define TIME  32
#define BSZ   10000

// ---------------- float<->ordered-uint maps (for atomic max of signed floats)
__device__ inline unsigned fmap(float f) {
  unsigned u = __float_as_uint(f);
  return (u & 0x80000000u) ? ~u : (u | 0x80000000u);
}
__device__ inline float funmap(unsigned u) {
  return (u & 0x80000000u) ? __uint_as_float(u & 0x7FFFFFFFu)
                           : __uint_as_float(~u);
}

__device__ inline float block_reduce(float v, float* sh) {
  int t = threadIdx.x;
  __syncthreads();
  sh[t] = v;
  __syncthreads();
  for (int s = 128; s > 0; s >>= 1) {
    if (t < s) sh[t] += sh[t + s];
    __syncthreads();
  }
  return sh[0];
}

__device__ inline v16h cat8(v8h lo, v8h hi) {
  return __builtin_shufflevector(lo, hi, 0, 1, 2, 3, 4, 5, 6, 7,
                                 8, 9, 10, 11, 12, 13, 14, 15);
}

// ---------------------------------------------------------------------------
// WMMA GEMM, K = 128 fixed:  C[M,N] = act(A[M,128] @ Wt^T + bias (+f32 tail))
// A: f16 [M,128] row-major.  Wt: f16 [N,128] (transposed weight: Wt[n][k]).
// Per the ISA 16-bit A layout, a lane's fragment = two contiguous K-runs of 8
// halves -> two 16B loads each for A and B. N multiple of 64; one wave owns a
// 16x64 tile. Tail tile (M%16) uses clamped loads + guarded stores.
// ---------------------------------------------------------------------------
template <bool RELU, bool HAS_TAIL, typename OutT>
__global__ void wmma_gemm(const f16* __restrict__ A, const f16* __restrict__ Wt,
                          const float* __restrict__ bias,
                          const float* __restrict__ A2, const float* __restrict__ W2, int K2,
                          OutT* __restrict__ C, int M, int N) {
  constexpr int K = 128;
  const int lane  = threadIdx.x & 31;
  const int wave  = threadIdx.x >> 5;
  const int tileN = blockIdx.x * 64;
  const int tileM = (blockIdx.y * 8 + wave) * 16;
  if (tileM >= M) return;                 // wave-uniform: EXEC stays all-ones
  const int  r = lane & 15, khalf = lane >> 4;
  const bool full = (tileM + 16 <= M);    // wave-uniform
  const int  rowc = min(tileM + r, M - 1);
  const f16* ap = A + (long)rowc * K;
  const f16* wp0 = Wt + (long)(tileN + r) * K;

  v8f acc0 = {}, acc1 = {}, acc2 = {}, acc3 = {};
#pragma unroll
  for (int kk = 0; kk < K; kk += 32) {
    const int kb0 = kk + khalf * 8;      // 16B-aligned half offsets
    const int kb1 = kk + 16 + khalf * 8;
    const v16h a = cat8(*(const v8h*)(ap + kb0), *(const v8h*)(ap + kb1));
#pragma unroll
    for (int nt = 0; nt < 4; ++nt) {
      const f16* wp = wp0 + (long)nt * 16 * K;
      const v16h b = cat8(*(const v8h*)(wp + kb0), *(const v8h*)(wp + kb1));
      if (nt == 0)      acc0 = __builtin_amdgcn_wmma_f32_16x16x32_f16(false, a, false, b, (short)0, acc0, false, false);
      else if (nt == 1) acc1 = __builtin_amdgcn_wmma_f32_16x16x32_f16(false, a, false, b, (short)0, acc1, false, false);
      else if (nt == 2) acc2 = __builtin_amdgcn_wmma_f32_16x16x32_f16(false, a, false, b, (short)0, acc2, false, false);
      else              acc3 = __builtin_amdgcn_wmma_f32_16x16x32_f16(false, a, false, b, (short)0, acc3, false, false);
    }
  }

#pragma unroll
  for (int nt = 0; nt < 4; ++nt) {
    const v8f& acc = (nt == 0) ? acc0 : (nt == 1) ? acc1 : (nt == 2) ? acc2 : acc3;
    const int col  = tileN + nt * 16 + r;
    const float bcol = bias[col];
    if (full) {
#pragma unroll
      for (int j = 0; j < 8; ++j) {
        const int row = tileM + j + (khalf ? 8 : 0);
        float val = acc[j] + bcol;
        if (HAS_TAIL) {
          const float* a2 = A2 + (long)row * K2;
          for (int u = 0; u < K2; ++u) val += a2[u] * W2[(long)u * N + col];
        }
        if (RELU) val = fmaxf(val, 0.f);
        C[(long)row * N + col] = (OutT)val;
      }
    } else {
#pragma unroll
      for (int j = 0; j < 8; ++j) {
        const int row = tileM + j + (khalf ? 8 : 0);
        if (row >= M) continue;
        float val = acc[j] + bcol;
        if (HAS_TAIL) {
          const float* a2 = A2 + (long)row * K2;
          for (int u = 0; u < K2; ++u) val += a2[u] * W2[(long)u * N + col];
        }
        if (RELU) val = fmaxf(val, 0.f);
        C[(long)row * N + col] = (OutT)val;
      }
    }
  }
}

// ---------------------------------------------------------------------------
// Edge time-encoding GEMM: ee[e,n] = sum_t cos(dts[e]*freq[t]+phase[t])*we[t,n]
// K = TIME = 32 -> one WMMA per column tile. A computed on the fly; B from
// pre-transposed f16 wet [128,32] -> two 16B loads per fragment.
// ---------------------------------------------------------------------------
__global__ void ee_gemm(const float* __restrict__ dts, const float* __restrict__ freq,
                        const float* __restrict__ phase, const f16* __restrict__ wet,
                        f16* __restrict__ ee, int E) {
  const int lane  = threadIdx.x & 31;
  const int wave  = threadIdx.x >> 5;
  const int tileM = (blockIdx.x * 8 + wave) * 16;
  if (tileM >= E) return;
  const int  r = lane & 15, khalf = lane >> 4;
  const bool full = (tileM + 16 <= E);
  const int  row  = min(tileM + r, E - 1);   // clamped; tail stores guarded
  const float t = dts[row];

  v16h a;
#pragma unroll
  for (int i = 0; i < 8; ++i) {
    const int kL = khalf * 8 + i;        // K-run 0: k = khalf*8 + 0..7
    const int kH = 16 + khalf * 8 + i;   // K-run 1: k = 16 + khalf*8 + 0..7
    a[i]     = (f16)__cosf(t * freq[kL] + phase[kL]);
    a[8 + i] = (f16)__cosf(t * freq[kH] + phase[kH]);
  }
#pragma unroll
  for (int nt = 0; nt < 8; ++nt) {
    const int col = nt * 16 + r;
    const f16* wp = wet + (long)col * TIME;
    const v16h b = cat8(*(const v8h*)(wp + khalf * 8),
                        *(const v8h*)(wp + 16 + khalf * 8));
    v8f acc = {};
    acc = __builtin_amdgcn_wmma_f32_16x16x32_f16(false, a, false, b,
                                                 (short)0, acc, false, false);
    if (full) {
#pragma unroll
      for (int j = 0; j < 8; ++j) {
        const int rr = tileM + j + (khalf ? 8 : 0);
        ee[(long)rr * HID + col] = (f16)acc[j];
      }
    } else {
#pragma unroll
      for (int j = 0; j < 8; ++j) {
        const int rr = tileM + j + (khalf ? 8 : 0);
        if (rr < E) ee[(long)rr * HID + col] = (f16)acc[j];
      }
    }
  }
}

// ---------------------------------------------------------------------------
// Operand preparation
// ---------------------------------------------------------------------------
// Wt[n*K + k] = (f16) W[k*N + n]   (transpose + f16 convert; tiny matrices)
__global__ void convert_wt(const float* __restrict__ W, f16* __restrict__ Wt,
                           int K, int N) {
  int tid = blockIdx.x * 256 + threadIdx.x;
  if (tid >= N * K) return;
  int col = tid / K, k = tid % K;
  Wt[tid] = (f16)W[(long)k * N + col];
}

// out[i,:] = (f16) x_all[ids[i],:]
__global__ void gather_f16(const float* __restrict__ x_all, const int* __restrict__ ids,
                           f16* __restrict__ out, int n) {
  long tid = (long)blockIdx.x * 256 + threadIdx.x;   // 4 elems per thread
  if (tid >= (long)n * 32) return;
  int row = (int)(tid >> 5), c4 = (int)(tid & 31) * 4;
  const float4 v = *(const float4*)(x_all + (long)ids[row] * HID + c4);
  f16* o = out + (long)row * HID + c4;
  o[0] = (f16)v.x; o[1] = (f16)v.y; o[2] = (f16)v.z; o[3] = (f16)v.w;
}

// ---------------------------------------------------------------------------
// Feature augmentation
// ---------------------------------------------------------------------------
__global__ void init_stats(float* outdeg, float* indeg, unsigned* mnb, unsigned* mxb,
                           float* sumd, float* sumsq, float* burst, float* seedacc, int n) {
  int i = blockIdx.x * 256 + threadIdx.x;
  if (i < n) {
    outdeg[i] = 0.f; indeg[i] = 0.f;
    mnb[i] = __float_as_uint(1.0e9f);   // min(1e9, segmin)
    mxb[i] = 0u;                        // max(0, segmax); dts >= 0
    sumd[i] = 0.f; sumsq[i] = 0.f; burst[i] = 0.f;
  }
  if (i < 16) seedacc[i] = 0.f;
}

__global__ void edge_stats(const int* __restrict__ src, const int* __restrict__ dst,
                           const float* __restrict__ dts, int E,
                           float* outdeg, float* indeg, unsigned* mnb, unsigned* mxb,
                           float* sumd, float* sumsq) {
  int e = blockIdx.x * 256 + threadIdx.x;
  if (e >= E) return;
  int s = src[e], d = dst[e];
  float t = dts[e];
  atomicAdd(outdeg + s, 1.f);
  atomicAdd(indeg + d, 1.f);
  atomicMin(mnb + d, __float_as_uint(t));   // valid: t >= 0
  atomicMax(mxb + d, __float_as_uint(t));
  atomicAdd(sumd + d, t);
  atomicAdd(sumsq + d, t * t);
}

__global__ void edge_burst(const int* __restrict__ dst, const float* __restrict__ dts, int E,
                           const unsigned* __restrict__ mnb, const unsigned* __restrict__ mxb,
                           float* burst) {
  int e = blockIdx.x * 256 + threadIdx.x;
  if (e >= E) return;
  int d = dst[e];
  float mn = __uint_as_float(mnb[d]);
  float mx = __uint_as_float(mxb[d]);
  float aw = fmaxf(mx - mn, 1.f);
  float cutoff = mn + 0.25f * aw;
  if (dts[e] <= cutoff) atomicAdd(burst + d, 1.f);
}

__global__ void node_extra_k(const float* __restrict__ outdeg, const float* __restrict__ indeg,
                             const unsigned* __restrict__ mnb, const unsigned* __restrict__ mxb,
                             const float* __restrict__ sumd, const float* __restrict__ sumsq,
                             const float* __restrict__ burst,
                             float* __restrict__ extra, float* seedacc, int n, int bs) {
  __shared__ float sh[256];
  int i = blockIdx.x * 256 + threadIdx.x;
  float f[7] = {0.f, 0.f, 0.f, 0.f, 0.f, 0.f, 0.f};
  if (i < n) {
    float od = outdeg[i], idg = indeg[i];
    float mn = __uint_as_float(mnb[i]);
    float cnt = fmaxf(idg, 1.f);
    float mean = sumd[i] / cnt;
    float msq  = sumsq[i] / cnt;
    f[0] = od;
    f[1] = idg;
    f[2] = od / (idg + 1.f);
    f[3] = fminf(mn, 1.0e8f);           // recency
    f[4] = burst[i] / cnt;              // burst_ratio
    f[5] = mean;
    f[6] = sqrtf(fmaxf(msq - mean * mean, 0.f));
#pragma unroll
    for (int j = 0; j < 7; ++j) extra[(long)i * 7 + j] = f[j];
  }
  bool seed = (i < bs) && (i < n);
#pragma unroll
  for (int j = 0; j < 7; ++j) {
    float s = block_reduce(seed ? f[j] : 0.f, sh);
    if (threadIdx.x == 0) atomicAdd(seedacc + j, s);
    float q = block_reduce(seed ? f[j] * f[j] : 0.f, sh);
    if (threadIdx.x == 0) atomicAdd(seedacc + 7 + j, q);
  }
}

__global__ void fin_seed(float* acc, int bs) {
  int j = threadIdx.x;
  if (j < 7) {
    float s = acc[j], s2 = acc[7 + j];
    float mean = s / bs;
    float var = (s2 - bs * mean * mean) / (float)(bs - 1);  // ddof=1
    acc[j] = mean;
    acc[7 + j] = fmaxf(sqrtf(fmaxf(var, 0.f)), 1e-8f);
  }
}

__global__ void norm_extra(float* extra, const float* __restrict__ seed, int n) {
  long tid = (long)blockIdx.x * 256 + threadIdx.x;
  if (tid >= (long)n * 7) return;
  int j = (int)(tid % 7);
  extra[tid] = (extra[tid] - seed[j]) / seed[7 + j];
}

// ---------------------------------------------------------------------------
// Edge attention
// ---------------------------------------------------------------------------
__global__ void edge_logits_k(const int* __restrict__ src, const int* __restrict__ dst,
                              const float* __restrict__ q, const float* __restrict__ k,
                              const f16* __restrict__ ee, int E,
                              float* __restrict__ logits, unsigned* __restrict__ mbuf) {
  int tid = blockIdx.x * 256 + threadIdx.x;
  if (tid >= E * HEADS) return;
  int e = tid >> 2, head = tid & 3;
  int s = src[e], d = dst[e];
  const float* qd = q + (long)d * HID + head * CH;
  const float* ks = k + (long)s * HID + head * CH;
  const f16*   ep = ee + (long)e * HID + head * CH;
  float acc = 0.f;
#pragma unroll
  for (int c = 0; c < CH; ++c) acc += qd[c] * (ks[c] + (float)ep[c]);
  float logit = acc * 0.17677669529663689f;  // 1/sqrt(32)
  logits[tid] = logit;
  atomicMax(mbuf + (long)d * HEADS + head, fmap(logit));
}

__global__ void edge_exp_den_k(const int* __restrict__ dst, int E,
                               float* __restrict__ logits,
                               const unsigned* __restrict__ mbuf,
                               float* __restrict__ den) {
  int tid = blockIdx.x * 256 + threadIdx.x;
  if (tid >= E * HEADS) return;
  int e = tid >> 2, head = tid & 3;
  int d = dst[e];
  float mm = funmap(mbuf[(long)d * HEADS + head]);
  float ex = __expf(logits[tid] - mm);
  logits[tid] = ex;                       // overwrite with exp
  atomicAdd(den + (long)d * HEADS + head, ex);
}

__global__ void edge_msg_k(const int* __restrict__ src, const int* __restrict__ dst,
                           const float* __restrict__ v, const f16* __restrict__ ee,
                           const float* __restrict__ exbuf, const float* __restrict__ den,
                           float* __restrict__ out, int E) {
  long tid = (long)blockIdx.x * 256 + threadIdx.x;   // one thread = 4 channels
  if (tid >= (long)E * 32) return;
  int e  = (int)(tid >> 5);
  int c4 = (int)(tid & 31) * 4;
  int head = c4 >> 5;
  int s = src[e], d = dst[e];
  float alpha = exbuf[(long)e * HEADS + head] /
                (den[(long)d * HEADS + head] + 1e-16f);
  const float4 vv = *(const float4*)(v + (long)s * HID + c4);   // 16B aligned
  const f16* ep = ee + (long)e * HID + c4;                      // 8B aligned
  float* op = out + (long)d * HID + c4;
  atomicAdd(op + 0, (vv.x + (float)ep[0]) * alpha);
  atomicAdd(op + 1, (vv.y + (float)ep[1]) * alpha);
  atomicAdd(op + 2, (vv.z + (float)ep[2]) * alpha);
  atomicAdd(op + 3, (vv.w + (float)ep[3]) * alpha);
}

// ---------------------------------------------------------------------------
// Batchnorm (column stats over M rows, then apply)
// ---------------------------------------------------------------------------
template <bool RELU_IN>
__global__ void bn_stats(const float* __restrict__ X, int M, int N,
                         float* sum, float* sumsq) {
  const int ROWS = 1024;
  int col = threadIdx.x % N;
  int sub = threadIdx.x / N;
  int per = 256 / N;
  int row0 = blockIdx.x * ROWS;
  int rowEnd = min(row0 + ROWS, M);
  float s = 0.f, s2 = 0.f;
  for (int r = row0 + sub; r < rowEnd; r += per) {
    float x = X[(long)r * N + col];
    if (RELU_IN) x = fmaxf(x, 0.f);
    s += x; s2 += x * x;
  }
  __shared__ float sh[256], sh2[256];
  sh[threadIdx.x] = s; sh2[threadIdx.x] = s2;
  __syncthreads();
  if (sub == 0) {
    for (int p = 1; p < per; ++p) { s += sh[p * N + col]; s2 += sh2[p * N + col]; }
    atomicAdd(sum + col, s);
    atomicAdd(sumsq + col, s2);
  }
}

template <bool RELU_IN, bool RELU_OUT, typename OutT>
__global__ void bn_apply(const float* __restrict__ X, OutT* __restrict__ Y, int M, int N,
                         const float* __restrict__ sum, const float* __restrict__ sumsq,
                         const float* __restrict__ g, const float* __restrict__ b) {
  long tid = (long)blockIdx.x * 256 + threadIdx.x;
  if (tid >= (long)M * N) return;
  int col = (int)(tid % N);
  float inv = 1.f / (float)M;
  float mu  = sum[col] * inv;
  float var = sumsq[col] * inv - mu * mu;   // ddof=0 (jnp .var default)
  float x = X[tid];
  if (RELU_IN) x = fmaxf(x, 0.f);
  float y = g[col] * (x - mu) * rsqrtf(var + 1e-5f) + b[col];
  if (RELU_OUT) y = fmaxf(y, 0.f);
  Y[tid] = (OutT)y;
}

__global__ void final_dot(const float* __restrict__ Z, const float* __restrict__ w3,
                          const float* __restrict__ b3, float* __restrict__ out, int M) {
  int r = blockIdx.x * 256 + threadIdx.x;
  if (r >= M) return;
  float acc = b3[0];
#pragma unroll
  for (int c = 0; c < 64; ++c) acc += Z[(long)r * 64 + c] * w3[c];
  out[r] = acc;
}

// ---------------------------------------------------------------------------
// Host orchestration
// ---------------------------------------------------------------------------
static inline int ceildiv(long a, long b) { return (int)((a + b - 1) / b); }

extern "C" void kernel_launch(void* const* d_in, const int* in_sizes, int n_in,
                              void* d_out, int out_size, void* d_ws, size_t ws_size,
                              hipStream_t stream) {
  // ---- inputs (setup_inputs dict order; params flattened in JAX pytree
  //      order = dict keys sorted alphabetically at every level) ----
  const float* x_all    = (const float*)d_in[0];   // (500000,128)
  const int*   node_ids = (const int*)  d_in[1];   // (50000,)
  const int*   edge_idx = (const int*)  d_in[2];   // (2,E)
  const float* dts      = (const float*)d_in[3];   // (E,)
  // d_in[4] = batch_size scalar (BSZ=10000, known at compile time)
  const float* basis_freq = (const float*)d_in[5];
  const float* clf_b1  = (const float*)d_in[6];
  const float* clf_b2  = (const float*)d_in[7];
  const float* clf_b3  = (const float*)d_in[8];
  const float* clf_bb1 = (const float*)d_in[9];
  const float* clf_bb2 = (const float*)d_in[10];
  const float* clf_g1  = (const float*)d_in[11];
  const float* clf_g2  = (const float*)d_in[12];
  const float* clf_w1  = (const float*)d_in[13];
  const float* clf_w2  = (const float*)d_in[14];
  const float* clf_w3  = (const float*)d_in[15];
  struct ConvP { const float *bk,*bn_b,*bn_g,*bq,*bskip,*bv,*we,*wk,*wq,*wskip,*wv; } cv[2];
  for (int L = 0; L < 2; ++L) {
    int base = 16 + L * 11;
    cv[L].bk    = (const float*)d_in[base + 0];
    cv[L].bn_b  = (const float*)d_in[base + 1];
    cv[L].bn_g  = (const float*)d_in[base + 2];
    cv[L].bq    = (const float*)d_in[base + 3];
    cv[L].bskip = (const float*)d_in[base + 4];
    cv[L].bv    = (const float*)d_in[base + 5];
    cv[L].we    = (const float*)d_in[base + 6];
    cv[L].wk    = (const float*)d_in[base + 7];
    cv[L].wq    = (const float*)d_in[base + 8];
    cv[L].wskip = (const float*)d_in[base + 9];
    cv[L].wv    = (const float*)d_in[base + 10];
  }
  const float* in_b  = (const float*)d_in[38];
  const float* in_w  = (const float*)d_in[39];   // (135,128) row-major
  const float* phase = (const float*)d_in[40];

  const int n = in_sizes[1];                     // 50000 sub-nodes
  const int E = in_sizes[3];                     // 600000 edges
  const int* src = edge_idx;
  const int* dst = edge_idx + E;

  // ---- workspace carving ----
  char* wsp = (char*)d_ws;
  auto carve = [&](size_t bytes) -> void* {
    void* p = (void*)wsp;
    wsp += (bytes + 255) & ~(size_t)255;
    return p;
  };
  f16*      h16    = (f16*)     carve((size_t)n * HID * 2);   // activations f16
  f16*      x16    = (f16*)     carve((size_t)n * HID * 2);   // gathered input / clf z1
  float*    qb     = (float*)   carve((size_t)n * HID * 4);
  float*    kb     = (float*)   carve((size_t)n * HID * 4);
  float*    vb     = (float*)   carve((size_t)n * HID * 4);
  float*    ob     = (float*)   carve((size_t)n * HID * 4);
  f16*      ee     = (f16*)     carve((size_t)E * HID * 2);
  float*    logits = (float*)   carve((size_t)E * HEADS * 4);
  unsigned* mbuf   = (unsigned*)carve((size_t)n * HEADS * 4);
  float*    den    = (float*)   carve((size_t)n * HEADS * 4);
  float*    extra  = (float*)   carve((size_t)n * 7 * 4);
  float*    outdeg = (float*)   carve((size_t)n * 4);
  float*    indeg  = (float*)   carve((size_t)n * 4);
  unsigned* mnb    = (unsigned*)carve((size_t)n * 4);
  unsigned* mxb    = (unsigned*)carve((size_t)n * 4);
  float*    sumd   = (float*)   carve((size_t)n * 4);
  float*    sumsq  = (float*)   carve((size_t)n * 4);
  float*    burst  = (float*)   carve((size_t)n * 4);
  float*    seedac = (float*)   carve(16 * 4);
  float*    bnsum  = (float*)   carve(HID * 4);
  float*    bnsq   = (float*)   carve(HID * 4);
  f16*      wt     = (f16*)     carve((size_t)HID * HID * 2); // transposed weight scratch
  f16*      wet    = (f16*)     carve((size_t)HID * TIME * 2);

  const dim3 blk(256);

  // ---- feature augmentation ----
  init_stats<<<ceildiv(n, 256), blk, 0, stream>>>(outdeg, indeg, mnb, mxb, sumd,
                                                  sumsq, burst, seedac, n);
  edge_stats<<<ceildiv(E, 256), blk, 0, stream>>>(src, dst, dts, E, outdeg, indeg,
                                                  mnb, mxb, sumd, sumsq);
  edge_burst<<<ceildiv(E, 256), blk, 0, stream>>>(dst, dts, E, mnb, mxb, burst);
  node_extra_k<<<ceildiv(n, 256), blk, 0, stream>>>(outdeg, indeg, mnb, mxb, sumd,
                                                    sumsq, burst, extra, seedac, n, BSZ);
  fin_seed<<<1, 32, 0, stream>>>(seedac, BSZ);
  norm_extra<<<ceildiv((long)n * 7, 256), blk, 0, stream>>>(extra, seedac, n);

  // ---- input projection: h = relu([x_all[node_ids] | extra] @ in_w + in_b) ----
  gather_f16<<<ceildiv((long)n * 32, 256), blk, 0, stream>>>(x_all, node_ids, x16, n);
  {
    convert_wt<<<ceildiv(HID * HID, 256), blk, 0, stream>>>(in_w, wt, HID, HID);
    dim3 g(HID / 64, ceildiv(ceildiv(n, 16), 8));
    wmma_gemm<true, true, f16><<<g, blk, 0, stream>>>(
        x16, wt, in_b, extra, in_w + 128 * HID, 7, h16, n, HID);
  }

  // ---- transformer conv layers ----
  for (int L = 0; L < 2; ++L) {
    dim3 g(HID / 64, ceildiv(ceildiv(n, 16), 8));
    convert_wt<<<ceildiv(HID * HID, 256), blk, 0, stream>>>(cv[L].wq, wt, HID, HID);
    wmma_gemm<false, false, float><<<g, blk, 0, stream>>>(
        h16, wt, cv[L].bq, nullptr, nullptr, 0, qb, n, HID);
    convert_wt<<<ceildiv(HID * HID, 256), blk, 0, stream>>>(cv[L].wk, wt, HID, HID);
    wmma_gemm<false, false, float><<<g, blk, 0, stream>>>(
        h16, wt, cv[L].bk, nullptr, nullptr, 0, kb, n, HID);
    convert_wt<<<ceildiv(HID * HID, 256), blk, 0, stream>>>(cv[L].wv, wt, HID, HID);
    wmma_gemm<false, false, float><<<g, blk, 0, stream>>>(
        h16, wt, cv[L].bv, nullptr, nullptr, 0, vb, n, HID);
    convert_wt<<<ceildiv(HID * HID, 256), blk, 0, stream>>>(cv[L].wskip, wt, HID, HID);
    wmma_gemm<false, false, float><<<g, blk, 0, stream>>>(
        h16, wt, cv[L].bskip, nullptr, nullptr, 0, ob, n, HID);

    convert_wt<<<ceildiv(HID * TIME, 256), blk, 0, stream>>>(cv[L].we, wet, TIME, HID);
    ee_gemm<<<ceildiv(ceildiv(E, 16), 8), blk, 0, stream>>>(
        dts, basis_freq, phase, wet, ee, E);

    hipMemsetAsync(mbuf, 0, (size_t)n * HEADS * 4, stream);  // fmap identity
    hipMemsetAsync(den,  0, (size_t)n * HEADS * 4, stream);
    edge_logits_k<<<ceildiv((long)E * HEADS, 256), blk, 0, stream>>>(
        src, dst, qb, kb, ee, E, logits, mbuf);
    edge_exp_den_k<<<ceildiv((long)E * HEADS, 256), blk, 0, stream>>>(
        dst, E, logits, mbuf, den);
    edge_msg_k<<<ceildiv((long)E * 32, 256), blk, 0, stream>>>(
        src, dst, vb, ee, logits, den, ob, E);

    hipMemsetAsync(bnsum, 0, HID * 4, stream);
    hipMemsetAsync(bnsq,  0, HID * 4, stream);
    bn_stats<true><<<ceildiv(n, 1024), blk, 0, stream>>>(ob, n, HID, bnsum, bnsq);
    bn_apply<true, false, f16><<<ceildiv((long)n * HID, 256), blk, 0, stream>>>(
        ob, h16, n, HID, bnsum, bnsq, cv[L].bn_g, cv[L].bn_b);
  }

  // ---- classifier head on first BSZ rows ----
  {
    convert_wt<<<ceildiv(HID * HID, 256), blk, 0, stream>>>(clf_w1, wt, HID, HID);
    dim3 g1(HID / 64, ceildiv(ceildiv(BSZ, 16), 8));
    wmma_gemm<false, false, float><<<g1, blk, 0, stream>>>(
        h16, wt, clf_b1, nullptr, nullptr, 0, qb, BSZ, HID);
    hipMemsetAsync(bnsum, 0, HID * 4, stream);
    hipMemsetAsync(bnsq,  0, HID * 4, stream);
    bn_stats<false><<<ceildiv(BSZ, 1024), blk, 0, stream>>>(qb, BSZ, HID, bnsum, bnsq);
    bn_apply<false, true, f16><<<ceildiv((long)BSZ * HID, 256), blk, 0, stream>>>(
        qb, x16, BSZ, HID, bnsum, bnsq, clf_g1, clf_bb1);   // z1 -> f16

    convert_wt<<<ceildiv(HID * 64, 256), blk, 0, stream>>>(clf_w2, wt, HID, 64);
    dim3 g2(1, ceildiv(ceildiv(BSZ, 16), 8));               // N = 64
    wmma_gemm<false, false, float><<<g2, blk, 0, stream>>>(
        x16, wt, clf_b2, nullptr, nullptr, 0, kb, BSZ, 64);
    hipMemsetAsync(bnsum, 0, HID * 4, stream);
    hipMemsetAsync(bnsq,  0, HID * 4, stream);
    bn_stats<false><<<ceildiv(BSZ, 1024), blk, 0, stream>>>(kb, BSZ, 64, bnsum, bnsq);
    bn_apply<false, true, float><<<ceildiv((long)BSZ * 64, 256), blk, 0, stream>>>(
        kb, kb, BSZ, 64, bnsum, bnsq, clf_g2, clf_bb2);

    final_dot<<<ceildiv(BSZ, 256), blk, 0, stream>>>(kb, clf_w3, clf_b3,
                                                     (float*)d_out, BSZ);
  }
  (void)n_in; (void)out_size; (void)ws_size;
}
